// BinaryConvBNReLU_89429809037363
// MI455X (gfx1250) — 8-bit matrix operand order repaired and checked by reading, compiled, not run on hardware
//
#include <hip/hip_runtime.h>
#include <stdint.h>

typedef __attribute__((ext_vector_type(8))) int v8i;

#define Bsz 32
#define Cch 256
#define Hh 56
#define Ww 56
#define HP 58
#define WP 58
#define KK 3
#define NW (Cch*Cch*KK*KK)
#define M_TOT (Bsz*Hh*Ww)
#define MTILES (M_TOT/16)
#define EPSV 1e-5f

__global__ void prep_w_kernel(const float* __restrict__ w, int* __restrict__ bprep) {
  int idx = blockIdx.x * blockDim.x + threadIdx.x;
  if (idx >= 9 * 4 * 16 * 32 * 8) return;
  int v     =  idx        & 7;
  int lane  = (idx >> 3)  & 31;
  int ntile = (idx >> 8)  & 15;
  int chunk = (idx >> 12) & 3;
  int tap   =  idx >> 14;
  int ky = tap / 3, kx = tap % 3;
  int cout = ntile * 16 + (lane & 15);
  unsigned int dword = 0;
  #pragma unroll
  for (int j = 0; j < 4; ++j) {
    int k   = 16 * (v >> 1) + ((lane & 16) ? 8 : 0) + (v & 1) * 4 + j;
    int cin = chunk * 64 + k;
    float wv = w[((cout * Cch + cin) * KK + ky) * KK + kx];
    int s = (wv > 0.f) ? 1 : ((wv < 0.f) ? -1 : 0);
    dword |= ((unsigned int)(unsigned char)(signed char)s) << (8 * j);
  }
  bprep[idx] = (int)dword;
}

__global__ void wmean_kernel(const float* __restrict__ w, float* __restrict__ meanval) {
  __shared__ float red[1024];
  float s = 0.f;
  for (int i = threadIdx.x; i < NW; i += 1024)
    s += fminf(fabsf(w[i]), 1.0f);
  red[threadIdx.x] = s;
  __syncthreads();
  for (int off = 512; off > 0; off >>= 1) {
    if (threadIdx.x < off) red[threadIdx.x] += red[threadIdx.x + off];
    __syncthreads();
  }
  if (threadIdx.x == 0) meanval[0] = red[0] / (float)NW;
}

__global__ void binx_kernel(const float* __restrict__ x, int* __restrict__ bxp) {
  int idx = blockIdx.x * blockDim.x + threadIdx.x;
  if (idx >= Bsz * HP * WP * Cch / 4) return;
  int c4  = idx & 63;
  int rem = idx >> 6;
  int xp = rem % WP; rem /= WP;
  int yp = rem % HP;
  int b  = rem / HP;
  unsigned int dword = 0;
  if (yp >= 1 && yp <= Hh && xp >= 1 && xp <= Ww) {
    int y = yp - 1, xx = xp - 1;
    #pragma unroll
    for (int j = 0; j < 4; ++j) {
      int c = c4 * 4 + j;
      float v = x[(((size_t)b * Cch + c) * Hh + y) * Ww + xx];
      int s = (v > 0.f) ? 1 : ((v < 0.f) ? -1 : 0);
      dword |= ((unsigned int)(unsigned char)(signed char)s) << (8 * j);
    }
  }
  bxp[idx] = (int)dword;
}

__global__ __launch_bounds__(512) void conv_wmma_kernel(
    const uint8_t* __restrict__ bxp, const int* __restrict__ bprep,
    const float* __restrict__ x, const float* __restrict__ meanval,
    float* __restrict__ pre, float* __restrict__ pSum, float* __restrict__ pSq) {
  const int lane  = threadIdx.x & 31;
  const int ntile = threadIdx.x >> 5;
  const int mtile = blockIdx.x;
  const int hi8   = (lane & 16) ? 8 : 0;

  int mg = mtile * 16 + (lane & 15);
  int xc = mg % Ww; int t = mg / Ww; int yc = t % Hh; int bi = t / Hh;

  v8i acc;
  #pragma unroll
  for (int i = 0; i < 8; ++i) acc[i] = 0;

  #pragma unroll
  for (int tap = 0; tap < 9; ++tap) {
    const int ky = tap / 3, kx = tap % 3;
    const uint8_t* pr =
        bxp + ((size_t)((bi * HP + yc + ky) * WP + (xc + kx))) * Cch + hi8;
    if (tap < 8) {
      const int kyn = (tap + 1) / 3, kxn = (tap + 1) % 3;
      __builtin_prefetch(
          bxp + ((size_t)((bi * HP + yc + kyn) * WP + (xc + kxn))) * Cch, 0, 0);
    }
    #pragma unroll
    for (int ch = 0; ch < 4; ++ch) {
      const uint8_t* pa = pr + ch * 64;
      uint2 q0 = *(const uint2*)(pa + 0);
      uint2 q1 = *(const uint2*)(pa + 16);
      uint2 q2 = *(const uint2*)(pa + 32);
      uint2 q3 = *(const uint2*)(pa + 48);
      v8i A; A[0]=(int)q0.x; A[1]=(int)q0.y; A[2]=(int)q1.x; A[3]=(int)q1.y;
             A[4]=(int)q2.x; A[5]=(int)q2.y; A[6]=(int)q3.x; A[7]=(int)q3.y;
      const int4* pb = (const int4*)(bprep +
          (((size_t)(tap * 4 + ch) * 16 + ntile) * 32 + lane) * 8);
      int4 b0 = pb[0], b1 = pb[1];
      v8i Bv; Bv[0]=b0.x; Bv[1]=b0.y; Bv[2]=b0.z; Bv[3]=b0.w;
              Bv[4]=b1.x; Bv[5]=b1.y; Bv[6]=b1.z; Bv[7]=b1.w;
      acc = __builtin_amdgcn_wmma_i32_16x16x64_iu8(
           true, A,  true, Bv, acc,
           false,  false);
    }
  }

  const float mv = meanval[0];
  const int cout = ntile * 16 + (lane & 15);
  float lsum = 0.f, lsq = 0.f;
  #pragma unroll
  for (int r = 0; r < 8; ++r) {
    int mrow = r + ((lane & 16) ? 8 : 0);
    int mg2 = mtile * 16 + mrow;
    int xx = mg2 % Ww; int t2 = mg2 / Ww; int yy = t2 % Hh; int bb = t2 / Hh;
    float val = mv * (float)acc[r] +
                x[(((size_t)bb * Cch + cout) * Hh + yy) * Ww + xx];
    pre[(size_t)mg2 * Cch + cout] = val;
    lsum += val; lsq += val * val;
  }
  lsum += __shfl_xor(lsum, 16, 32);
  lsq  += __shfl_xor(lsq, 16, 32);
  if (lane < 16) {
    pSum[(size_t)mtile * Cch + cout] = lsum;
    pSq [(size_t)mtile * Cch + cout] = lsq;
  }
}

__global__ void bn_stats_kernel(const float* __restrict__ pSum,
                                const float* __restrict__ pSq,
                                const float* __restrict__ gamma,
                                const float* __restrict__ beta,
                                float* __restrict__ scale,
                                float* __restrict__ shiftv) {
  int c = threadIdx.x;
  float s = 0.f, q = 0.f;
  for (int i = 0; i < MTILES; ++i) {
    s += pSum[(size_t)i * Cch + c];
    q += pSq [(size_t)i * Cch + c];
  }
  const float invN = 1.0f / (float)M_TOT;
  float mean = s * invN;
  float var  = q * invN - mean * mean;
  float inv  = rsqrtf(var + EPSV);
  float sc   = gamma[c] * inv;
  scale[c]  = sc;
  shiftv[c] = beta[c] - mean * sc;
}

__global__ void bn_apply_kernel(const float* __restrict__ pre,
                                const float* __restrict__ scale,
                                const float* __restrict__ shiftv,
                                float* __restrict__ out) {
  __shared__ float tile[16][17];
  int row = blockIdx.z;
  int b = row / Hh, y = row % Hh;
  int x0 = blockIdx.x * 16;
  int c0 = blockIdx.y * 16;
  int tx = threadIdx.x, ty = threadIdx.y;
  int xl = x0 + ty;
  if (xl < Ww)
    tile[ty][tx] = pre[((size_t)row * Ww + xl) * Cch + (c0 + tx)];
  __syncthreads();
  int xs = x0 + tx;
  int c  = c0 + ty;
  if (xs < Ww) {
    float v = tile[tx][ty] * scale[c] + shiftv[c];
    out[(((size_t)b * Cch + c) * Hh + y) * Ww + xs] = fmaxf(v, 0.f);
  }
}

extern "C" void kernel_launch(void* const* d_in, const int* in_sizes, int n_in,
                              void* d_out, int out_size, void* d_ws, size_t ws_size,
                              hipStream_t stream) {
  const float* x     = (const float*)d_in[0];
  const float* w     = (const float*)d_in[1];
  const float* gamma = (const float*)d_in[2];
  const float* beta  = (const float*)d_in[3];
  float* out = (float*)d_out;
  char*  ws  = (char*)d_ws;

  float* meanval = (float*)(ws + 0);
  float* scale   = (float*)(ws + 4096);
  float* shiftv  = (float*)(ws + 8192);
  int*   bprep   = (int*)  (ws + 16384);
  int*   bxp     = (int*)  (ws + (1ull << 20));
  float* pre     = (float*)(ws + 33554432ull);
  float* pSum    = (float*)(ws + 136314880ull);
  float* pSq     = (float*)(ws + 142737408ull);

  prep_w_kernel<<<(9 * 4 * 16 * 32 * 8) / 256, 256, 0, stream>>>(w, bprep);
  wmean_kernel<<<1, 1024, 0, stream>>>(w, meanval);
  binx_kernel<<<(Bsz * HP * WP * Cch / 4) / 256, 256, 0, stream>>>(x, bxp);
  conv_wmma_kernel<<<MTILES, 512, 0, stream>>>(
      (const uint8_t*)bxp, bprep, x, meanval, pre, pSum, pSq);
  bn_stats_kernel<<<1, 256, 0, stream>>>(pSum, pSq, gamma, beta, scale, shiftv);
  dim3 bt(16, 16, 1), gt((Ww + 15) / 16, Cch / 16, Bsz * Hh);
  bn_apply_kernel<<<gt, bt, 0, stream>>>(pre, scale, shiftv, out);
}
